// UltimateMoE_44925357916270
// MI455X (gfx1250) — matrix operand order repaired and checked by reading, not compiled, not run on hardware
//
#include <hip/hip_runtime.h>
#include <hip/hip_bf16.h>
#include <stdint.h>

#define NUM_EXPERTS 16
#define D_MODEL 1024
#define M_TOKENS 8192
#define TILE_M 64       // tokens per block: B-fragment reuse x8 -> ~0.5GB total B traffic
#define MAX_PER_EXPERT M_TOKENS
#define BLOCK_N 256
#define A_STRIDE 1032   // bf16 elems per A row (1024 + 8 pad): fragment reads hit distinct banks
#define SMEM_BYTES (TILE_M * A_STRIDE * 2 + TILE_M * 4 + TILE_M * 4)

typedef __attribute__((ext_vector_type(16))) __bf16 v16bf;
typedef __attribute__((ext_vector_type(8)))  float  v8f;
typedef __attribute__((ext_vector_type(4)))  int    v4i;

#if __has_builtin(__builtin_amdgcn_global_load_async_to_lds_b128)
#define USE_ASYNC_LDS 1
#endif

__device__ __forceinline__ unsigned int pk_bf16(float a, float b) {
  union { float f; unsigned int u; } ca, cb; ca.f = a; cb.f = b;
  unsigned int ua = ca.u + (0x7FFFu + ((ca.u >> 16) & 1u));   // RNE
  unsigned int ub = cb.u + (0x7FFFu + ((cb.u >> 16) & 1u));
  return (ua >> 16) | (ub & 0xFFFF0000u);
}

// ---- Stage 0a: x (f32 row-major) -> Xbf (bf16 row-major) -------------------
__global__ void __launch_bounds__(256) convert_x(const float* __restrict__ x,
                                                 unsigned short* __restrict__ xbf) {
  int i = blockIdx.x * blockDim.x + threadIdx.x;     // one float4 per thread
  float4 v = ((const float4*)x)[i];
  uint2 o; o.x = pk_bf16(v.x, v.y); o.y = pk_bf16(v.z, v.w);
  ((uint2*)xbf)[i] = o;
}

// ---- Stage 0b: W[e][k][n] f32 -> Wt[e][n][k] bf16 (LDS-tiled transpose) ----
__global__ void __launch_bounds__(256) convert_w(const float* __restrict__ w,
                                                 unsigned short* __restrict__ wt) {
  __shared__ float tile[64][65];
  const int e  = blockIdx.z;
  const int nt = blockIdx.x * 64;
  const int kt = blockIdx.y * 64;
  const int t  = threadIdx.x;
  const float* We = w + ((size_t)e << 20);
  unsigned short* Wte = wt + ((size_t)e << 20);

#pragma unroll
  for (int i = 0; i < 4; ++i) {                       // read: coalesced along n
    int k  = (t >> 4) + i * 16;
    int n4 = (t & 15) * 4;
    float4 v = *(const float4*)(We + (size_t)(kt + k) * D_MODEL + nt + n4);
    tile[k][n4 + 0] = v.x; tile[k][n4 + 1] = v.y;
    tile[k][n4 + 2] = v.z; tile[k][n4 + 3] = v.w;
  }
  __syncthreads();
#pragma unroll
  for (int i = 0; i < 4; ++i) {                       // write: coalesced along k
    int n  = (t >> 4) + i * 16;
    int k4 = (t & 15) * 4;
    uint2 o;
    o.x = pk_bf16(tile[k4 + 0][n], tile[k4 + 1][n]);
    o.y = pk_bf16(tile[k4 + 2][n], tile[k4 + 3][n]);
    *(uint2*)(Wte + (size_t)(nt + n) * D_MODEL + kt + k4) = o;
  }
}

// ---- Stage 1: gating (logits -> top2 -> softmax -> scatter) ----------------
__global__ void __launch_bounds__(256) moe_gate(
    const float* __restrict__ x, const float* __restrict__ gw,
    int* __restrict__ counts, int* __restrict__ tok_id, float* __restrict__ tok_w) {
  int m = blockIdx.x * blockDim.x + threadIdx.x;
  if (m >= M_TOKENS) return;

  const float4* xr = (const float4*)(x + (size_t)m * D_MODEL);
  float acc[NUM_EXPERTS];
#pragma unroll
  for (int e = 0; e < NUM_EXPERTS; ++e) acc[e] = 0.f;
  for (int k4 = 0; k4 < D_MODEL / 4; ++k4) {
    float4 xv = xr[k4];
#pragma unroll
    for (int e = 0; e < NUM_EXPERTS; ++e) {
      float4 g = ((const float4*)(gw + (size_t)e * D_MODEL))[k4];  // uniform -> broadcast
      acc[e] += xv.x * g.x + xv.y * g.y + xv.z * g.z + xv.w * g.w;
    }
  }
  int e1 = 0; float v1 = acc[0];
#pragma unroll
  for (int e = 1; e < NUM_EXPERTS; ++e) if (acc[e] > v1) { v1 = acc[e]; e1 = e; }
  int e2 = 0; float v2 = -__builtin_inff();
#pragma unroll
  for (int e = 0; e < NUM_EXPERTS; ++e) if (e != e1 && acc[e] > v2) { v2 = acc[e]; e2 = e; }

  float p  = __expf(v2 - v1);
  float w1 = 1.f / (1.f + p);
  float w2 = p * w1;

  int s1 = atomicAdd(&counts[e1], 1);
  tok_id[e1 * MAX_PER_EXPERT + s1] = m;
  tok_w [e1 * MAX_PER_EXPERT + s1] = w1;
  int s2 = atomicAdd(&counts[e2], 1);
  tok_id[e2 * MAX_PER_EXPERT + s2] = m;
  tok_w [e2 * MAX_PER_EXPERT + s2] = w2;
}

// ---- Stage 2: gathered per-expert GEMM, bf16 WMMA, B straight from L2 ------
// LDS-bound to 1 block/WGP anyway -> request max VGPRs (min waves/EU = 1).
__global__ void __launch_bounds__(256, 1) moe_expert_gemm(
    const unsigned short* __restrict__ xbf, const unsigned short* __restrict__ wt,
    const int* __restrict__ counts, const int* __restrict__ tok_id,
    const float* __restrict__ tok_w, float* __restrict__ out) {
  extern __shared__ __align__(16) char smem[];
  unsigned short* A_lds = (unsigned short*)smem;                       // 64 x 1032 bf16
  int*   s_tok = (int*)  (smem + TILE_M * A_STRIDE * 2);
  float* s_w   = (float*)(smem + TILE_M * A_STRIDE * 2 + TILE_M * 4);

  const int e  = blockIdx.y;
  const int mt = blockIdx.x;
  const int nb = blockIdx.z * BLOCK_N;
  const int cnt = counts[e];
  if (mt * TILE_M >= cnt) return;          // block-uniform: EXEC stays full

  const int tid   = threadIdx.x;
  const int lane  = tid & 31;
  const int wv    = tid >> 5;              // 8 waves, each owns N=32
  const int khalf = lane >> 4;
  const int nl    = lane & 15;

  if (tid < TILE_M) {
    int slot = mt * TILE_M + tid;
    if (slot < cnt) { s_tok[tid] = tok_id[e * MAX_PER_EXPERT + slot];
                      s_w[tid]   = tok_w [e * MAX_PER_EXPERT + slot]; }
    else            { s_tok[tid] = -1; s_w[tid] = 0.f; }
  }
  __syncthreads();

  // Stage A: gather 64 bf16 token rows (2KB each) -> LDS. 4 threads per row.
  {
    int r   = tid >> 2;                    // 0..63
    int seg = tid & 3;                     // 512B each
    int tk = s_tok[r]; if (tk < 0) tk = 0; // masked at epilogue
    const unsigned short* src = xbf + (size_t)tk * D_MODEL + seg * 256;
    unsigned short* dst = &A_lds[r * A_STRIDE + seg * 256];
#if defined(USE_ASYNC_LDS)
#pragma unroll
    for (int j = 0; j < 32; ++j)
      __builtin_amdgcn_global_load_async_to_lds_b128(
          (v4i*)(src + j * 8), (v4i*)(dst + j * 8), 0, 0);
    asm volatile("s_wait_asynccnt 0x0" ::: "memory");
#else
#pragma unroll
    for (int j = 0; j < 32; ++j)
      *(uint4*)(dst + j * 8) = *(const uint4*)(src + j * 8);
#endif
  }
  __syncthreads();

  // Per-lane pointers. B fragment = two 16-byte runs (k 8h..+7, 16+8h..+7) of pre-transposed Wt.
  const unsigned short* ap[4];
  ap[0] = &A_lds[nl * A_STRIDE + khalf * 8];
  ap[1] = ap[0] + 16 * A_STRIDE;
  ap[2] = ap[0] + 32 * A_STRIDE;
  ap[3] = ap[0] + 48 * A_STRIDE;
  const unsigned short* b0p = wt + ((size_t)e * D_MODEL + nb + wv * 32 + nl) * D_MODEL
                                 + khalf * 8;
  const unsigned short* b1p = b0p + 16 * D_MODEL;

  v8f acc[4][2];
#pragma unroll
  for (int a = 0; a < 4; ++a) { acc[a][0] = (v8f){}; acc[a][1] = (v8f){}; }

  // Double-buffered fragments: buffer `nxt` is loaded while `cur` feeds WMMA.
  union Frag { v16bf v; uint4 q[2]; };
  Frag af[2][4], bf[2][2];

  auto load_frags = [&](int buf, int k) {
    bf[buf][0].q[0] = *(const uint4*)(b0p + k);       // B: lane N=nl, K {8h..+7},{16+8h..+7}
    bf[buf][0].q[1] = *(const uint4*)(b0p + k + 16);
    bf[buf][1].q[0] = *(const uint4*)(b1p + k);
    bf[buf][1].q[1] = *(const uint4*)(b1p + k + 16);
#pragma unroll
    for (int a = 0; a < 4; ++a) {                     // A: lane M=nl(+16a), K {8h..+7},{16+8h..+7}
      af[buf][a].q[0] = *(const uint4*)(ap[a] + k);
      af[buf][a].q[1] = *(const uint4*)(ap[a] + k + 16);
    }
  };

  load_frags(0, 0);

#pragma unroll 2
  for (int k0 = 0; k0 < D_MODEL; k0 += 32) {
    const int cur = (k0 >> 5) & 1;
    const int nxt = cur ^ 1;
    const int kn  = (k0 + 32 < D_MODEL) ? (k0 + 32) : k0;  // last iter: harmless reload
    load_frags(nxt, kn);
#pragma unroll
    for (int a = 0; a < 4; ++a) {
      acc[a][0] = __builtin_amdgcn_wmma_f32_16x16x32_bf16(false, af[cur][a].v, false, bf[cur][0].v,
                                                          (short)0, acc[a][0], false, false);
      acc[a][1] = __builtin_amdgcn_wmma_f32_16x16x32_bf16(false, af[cur][a].v, false, bf[cur][1].v,
                                                          (short)0, acc[a][1], false, false);
    }
  }

  // Epilogue: C layout lane N=nl, VGPR j -> M = 16a + j + 8*khalf; scale + atomic combine.
  int n0 = nb + wv * 32 + nl;
  int n1 = n0 + 16;
#pragma unroll
  for (int a = 0; a < 4; ++a) {
#pragma unroll
    for (int j = 0; j < 8; ++j) {
      int m = a * 16 + j + khalf * 8;
      int t = s_tok[m];
      if (t >= 0) {
        float w = s_w[m];
        atomicAdd(&out[(size_t)t * D_MODEL + n0], w * acc[a][0][j]);
        atomicAdd(&out[(size_t)t * D_MODEL + n1], w * acc[a][1][j]);
      }
    }
  }
}

// ---- host launcher ----------------------------------------------------------
extern "C" void kernel_launch(void* const* d_in, const int* in_sizes, int n_in,
                              void* d_out, int out_size, void* d_ws, size_t ws_size,
                              hipStream_t stream) {
  const float* x  = (const float*)d_in[0];
  const float* gw = (const float*)d_in[1];
  const float* ew = (const float*)d_in[2];
  float* out = (float*)d_out;

  // ws: counts(64B) | tok_id 512KB | tok_w 512KB | Xbf 16MB | Wt 32MB  (~49MB)
  char* ws = (char*)d_ws;
  int*            counts = (int*)ws;
  int*            tok_id = (int*)(ws + 64);
  float*          tok_w  = (float*)(ws + 64 + (size_t)NUM_EXPERTS * MAX_PER_EXPERT * 4);
  unsigned short* xbf    = (unsigned short*)(ws + 64 + (size_t)NUM_EXPERTS * MAX_PER_EXPERT * 8);
  unsigned short* wt     = (unsigned short*)((char*)xbf + (size_t)M_TOKENS * D_MODEL * 2);

  (void)hipMemsetAsync(counts, 0, 64, stream);
  (void)hipMemsetAsync(d_out, 0, (size_t)M_TOKENS * D_MODEL * sizeof(float), stream);

  convert_x<<<dim3((M_TOKENS * D_MODEL / 4) / 256), dim3(256), 0, stream>>>(x, xbf);
  convert_w<<<dim3(D_MODEL / 64, D_MODEL / 64, NUM_EXPERTS), dim3(256), 0, stream>>>(ew, wt);
  moe_gate<<<dim3(M_TOKENS / 256), dim3(256), 0, stream>>>(x, gw, counts, tok_id, tok_w);

  dim3 grid(M_TOKENS / TILE_M, NUM_EXPERTS, D_MODEL / BLOCK_N);
  moe_expert_gemm<<<grid, dim3(256), SMEM_BYTES, stream>>>(xbf, wt, counts, tok_id, tok_w, out);
}
